// Network_3959959847635
// MI455X (gfx1250) — compile-verified, matrix operand order checked by reading, not run on hardware
//
#include <hip/hip_runtime.h>
#include <hip/hip_bf16.h>
#include <stdint.h>

// ---------------------------------------------------------------------------
// Types for CDNA5 WMMA (wave32, 16x16x32 f16 -> f32 accumulate)
// ---------------------------------------------------------------------------
typedef __attribute__((ext_vector_type(16))) _Float16 v16h;
typedef __attribute__((ext_vector_type(8)))  _Float16 v8h;
typedef __attribute__((ext_vector_type(8)))  float    v8f;
typedef __attribute__((ext_vector_type(4)))  float    f32x4;

// Problem constants (from setup_inputs): B=128, K=196, D=E=H=512, V=32000, T=20
#define BB   128
#define KK   196
#define DD   512
#define EE   512
#define HH   512
#define VV   32000
#define TT   20

// ---------------------------------------------------------------------------
// Fragment loaders for NT GEMM (contraction dim contiguous in memory).
// 16-bit A-matrix 16x32 layout (ISA 7.12.2): lanes 0-15 hold M=0..15 with
// K = {0..7, 16..23}; lanes 16-31 hold the same rows with K = {8..15, 24..31}.
// => per lane: two 8-element (16B) contiguous loads at k0+half*8 and k0+16+half*8.
// Dense B (32x16) mirrors this with N in place of M.
// ---------------------------------------------------------------------------
__device__ inline v16h load_frag(const _Float16* p, int half) {
  const v8h* a = reinterpret_cast<const v8h*>(p + half * 8);
  const v8h* b = reinterpret_cast<const v8h*>(p + 16 + half * 8);
  v8h x = *a, y = *b;
  v16h r;
#pragma unroll
  for (int i = 0; i < 8; ++i) { r[i] = x[i]; r[i + 8] = y[i]; }
  return r;
}

// fp32 source variant: load 8+8 floats and convert to f16 in-register
// (lets the KWk GEMM read `features` directly, no 25 MB staging copy).
__device__ inline v16h load_frag(const float* p, int half) {
  const f32x4* a = reinterpret_cast<const f32x4*>(p + half * 8);
  const f32x4* b = reinterpret_cast<const f32x4*>(p + 16 + half * 8);
  f32x4 x0 = a[0], x1 = a[1], y0 = b[0], y1 = b[1];
  v16h r;
#pragma unroll
  for (int i = 0; i < 4; ++i) {
    r[i]      = (_Float16)x0[i];
    r[4 + i]  = (_Float16)x1[i];
    r[8 + i]  = (_Float16)y0[i];
    r[12 + i] = (_Float16)y1[i];
  }
  return r;
}

// ---------------------------------------------------------------------------
// Async staging of one 64(N) x 64(K) f16 B-chunk into LDS via the gfx1250
// async LDS-DMA path (GLOBAL_LOAD_ASYNC_TO_LDS_B128, tracked by ASYNCcnt).
// 256 threads x 2 ops x 16B = 8KB. LDS offset = low 32 bits of flat shared
// pointer (LDS aperture: addr[31:0] is the wave-relative LDS byte offset).
// ---------------------------------------------------------------------------
__device__ inline void stage_chunk_async(const _Float16* __restrict__ Bbase, int Kd,
                                         _Float16 (*sB)[64][64], int buf, int c, int tid) {
#pragma unroll
  for (int rep = 0; rep < 2; ++rep) {
    int g  = tid + rep * 256;   // 0..511 groups of 8 f16 (16B)
    int n  = g >> 3;            // 0..63  (row of B strip)
    int kk = (g & 7) << 3;      // 0,8,...,56 (chunk-local k)
    const _Float16* gp = Bbase + (size_t)n * Kd + (size_t)c * 64 + kk;
    unsigned ldsoff = (unsigned)(uintptr_t)&sB[buf][n][kk];
    unsigned long long ga = (unsigned long long)(uintptr_t)gp;
    asm volatile("global_load_async_to_lds_b128 %0, %1, off"
                 :: "v"(ldsoff), "v"(ga) : "memory");
  }
}

// ---------------------------------------------------------------------------
// Block-cooperative NT GEMM: C[m][n] = sum_k A[m][k] * B[n][k] (+ bias[n]).
// Block (256 thr = 8 waves) computes a 128(M) x 64(N) strip; the shared
// 64x64 B chunk is async-staged into a double-buffered LDS tile and reused
// by all 8 waves (8x less L2 traffic on out_W than per-wave tiling).
// grid = (N/64, M/128). Requires M%128==0, N%64==0, Kd%64==0.
// ---------------------------------------------------------------------------
template <typename AT>
__global__ void gemm_nt_wmma_lds(const AT* __restrict__ A, const _Float16* __restrict__ B,
                                 const float* __restrict__ bias, float* __restrict__ C,
                                 int M, int N, int Kd) {
  __shared__ _Float16 sB[2][64][64];
  const int tid  = threadIdx.x;
  const int lane = tid & 31;
  const int wave = tid >> 5;
  const int row  = lane & 15;
  const int half = lane >> 4;
  const int m0 = blockIdx.y * 128 + wave * 16;
  const int n0 = blockIdx.x * 64;

  const _Float16* Bbase = B + (size_t)n0 * Kd;
  const AT*       arow  = A + (size_t)(m0 + row) * Kd;

  v8f acc[4] = { {0.f,0.f,0.f,0.f,0.f,0.f,0.f,0.f},
                 {0.f,0.f,0.f,0.f,0.f,0.f,0.f,0.f},
                 {0.f,0.f,0.f,0.f,0.f,0.f,0.f,0.f},
                 {0.f,0.f,0.f,0.f,0.f,0.f,0.f,0.f} };

  const int NC = Kd >> 6;  // 64-wide K chunks
  stage_chunk_async(Bbase, Kd, sB, 0, 0, tid);

  for (int c = 0; c < NC; ++c) {
    if (c + 1 < NC) {
      stage_chunk_async(Bbase, Kd, sB, (c + 1) & 1, c + 1, tid);
      // async loads complete in order: <=2 outstanding => chunk c has landed
      asm volatile("s_wait_asynccnt 0x2" ::: "memory");
    } else {
      asm volatile("s_wait_asynccnt 0x0" ::: "memory");
    }
    __syncthreads();

    const int buf = c & 1;
#pragma unroll
    for (int kc = 0; kc < 64; kc += 32) {
      v16h af = load_frag(arow + c * 64 + kc, half);
#pragma unroll
      for (int j = 0; j < 4; ++j) {
        v16h bfr = load_frag(&sB[buf][j * 16 + row][kc], half);  // ds_load_b128 x2
        acc[j] = __builtin_amdgcn_wmma_f32_16x16x32_f16(
            /*neg_a=*/false, af, /*neg_b=*/false, bfr,
            /*c_mod=*/(short)0, acc[j], /*reuse_a=*/false, /*reuse_b=*/false);
      }
    }
    __syncthreads();
  }

  // C/D layout: lane (0..15) -> N column, VGPR r -> M = r + 8*half.
#pragma unroll
  for (int j = 0; j < 4; ++j) {
    int n = n0 + j * 16 + row;
    float bv = bias ? bias[n] : 0.f;
#pragma unroll
    for (int r = 0; r < 8; ++r) {
      int m = m0 + r + 8 * half;
      C[(size_t)m * N + n] = acc[j][r] + bv;
    }
  }
}

// ---------------------------------------------------------------------------
// fp32 -> f16 conversion (grid stride)
// ---------------------------------------------------------------------------
__global__ void cvt_f16_kernel(const float* __restrict__ s, _Float16* __restrict__ d, size_t n) {
  size_t i = (size_t)blockIdx.x * blockDim.x + threadIdx.x;
  size_t stride = (size_t)gridDim.x * blockDim.x;
  for (; i < n; i += stride) d[i] = (_Float16)s[i];
}

// fp32 [R x C] -> f16 transposed [C x R] (for attn_Wk so the KWk GEMM is NT)
__global__ void cvt_f16_t_kernel(const float* __restrict__ s, _Float16* __restrict__ d,
                                 int R, int C) {
  int c = blockIdx.x * blockDim.x + threadIdx.x;
  int r = blockIdx.y * blockDim.y + threadIdx.y;
  if (r < R && c < C) d[(size_t)c * R + r] = (_Float16)s[(size_t)r * C + c];
}

__global__ void init_state_kernel(float* __restrict__ h, int* __restrict__ y, int nh, int nb) {
  int i = blockIdx.x * blockDim.x + threadIdx.x;
  if (i < nh) h[i] = 0.f;
  if (i < nb) y[i] = 1;  // SOS token
}

// ---------------------------------------------------------------------------
// Fused attention step, one block per batch element b:
//   q = h[b] @ Wq            (LDS)
//   score_k = sum_h relu(KWk[b,k,h] + q[h]) * v[h]
//   a = softmax(score)       (written to attn output slice)
//   ctx = sum_k a[k] * features[b,k,:]
//   x[b] = concat(emb_W[y[b]], ctx)
// ---------------------------------------------------------------------------
__global__ void attn_step_kernel(const float* __restrict__ h, const float* __restrict__ Wq,
                                 const float* __restrict__ KWk, const float* __restrict__ av,
                                 const float* __restrict__ features,
                                 const float* __restrict__ embW, const int* __restrict__ y,
                                 float* __restrict__ x, float* __restrict__ attn_out) {
  const int b = blockIdx.x, tid = threadIdx.x, bs = blockDim.x;
  __shared__ float s_h[HH];
  __shared__ float s_q[HH];
  __shared__ float s_sc[256];
  __shared__ float s_red[256];

  for (int i = tid; i < HH; i += bs) s_h[i] = h[(size_t)b * HH + i];
  __syncthreads();

  // q projection (h @ Wq), Wq is [H, H] row-major
  for (int j = tid; j < HH; j += bs) {
    float acc = 0.f;
    for (int k = 0; k < HH; ++k) acc += s_h[k] * Wq[(size_t)k * HH + j];
    s_q[j] = acc;
  }
  __syncthreads();

  // scores: wave per k, lane-strided over H, wave32 shuffle reduce
  const int wave = tid >> 5, lane = tid & 31, nwaves = bs >> 5;
  for (int k = wave; k < KK; k += nwaves) {
    const float* kw = KWk + ((size_t)b * KK + k) * HH;
    float s = 0.f;
    for (int hh = lane; hh < HH; hh += 32) {
      float t = kw[hh] + s_q[hh];
      s += fmaxf(t, 0.f) * av[hh];
    }
    for (int o = 16; o; o >>= 1) s += __shfl_xor(s, o, 32);
    if (lane == 0) s_sc[k] = s;
  }
  __syncthreads();

  // softmax over K=196
  float val = (tid < KK) ? s_sc[tid] : -3.4e38f;
  s_red[tid] = val;
  __syncthreads();
  for (int st = bs >> 1; st; st >>= 1) {
    if (tid < st) s_red[tid] = fmaxf(s_red[tid], s_red[tid + st]);
    __syncthreads();
  }
  float m = s_red[0];
  __syncthreads();
  float e = (tid < KK) ? expf(val - m) : 0.f;
  s_red[tid] = e;
  __syncthreads();
  for (int st = bs >> 1; st; st >>= 1) {
    if (tid < st) s_red[tid] += s_red[tid + st];
    __syncthreads();
  }
  float inv = 1.f / s_red[0];
  if (tid < KK) {
    float a = e * inv;
    s_sc[tid] = a;
    attn_out[(size_t)b * KK + tid] = a;
  }
  __syncthreads();

  // ctx = a @ features[b]
  for (int d = tid; d < DD; d += bs) {
    float c = 0.f;
    const float* f = features + (size_t)b * KK * DD + d;
    for (int k = 0; k < KK; ++k) c += s_sc[k] * f[(size_t)k * DD];
    x[(size_t)b * (EE + DD) + EE + d] = c;
  }
  // embedding gather
  int tok = y[b];
  for (int i = tid; i < EE; i += bs)
    x[(size_t)b * (EE + DD) + i] = embW[(size_t)tok * EE + i];
}

// ---------------------------------------------------------------------------
// GRU cell: one thread per (b, j). Also emits h_new in f16 for the logits GEMM.
// ---------------------------------------------------------------------------
__global__ void gru_step_kernel(const float* __restrict__ x, const float* __restrict__ h,
                                const float* __restrict__ W_ih, const float* __restrict__ W_hh,
                                const float* __restrict__ b_ih, const float* __restrict__ b_hh,
                                float* __restrict__ h_new, _Float16* __restrict__ h16) {
  const int XD = EE + DD;
  int idx = blockIdx.x * blockDim.x + threadIdx.x;  // b*H + j
  int b = idx >> 9, j = idx & (HH - 1);
  const float* xr = x + (size_t)b * XD;
  const float* hr = h + (size_t)b * HH;

  float ir = b_ih[j], iz = b_ih[HH + j], in_ = b_ih[2 * HH + j];
  const float* wr = W_ih + (size_t)j * XD;
  const float* wz = W_ih + (size_t)(HH + j) * XD;
  const float* wn = W_ih + (size_t)(2 * HH + j) * XD;
  for (int c = 0; c < XD; ++c) {
    float xv = xr[c];
    ir += xv * wr[c]; iz += xv * wz[c]; in_ += xv * wn[c];
  }
  float hrv = b_hh[j], hzv = b_hh[HH + j], hnv = b_hh[2 * HH + j];
  const float* vr = W_hh + (size_t)j * HH;
  const float* vz = W_hh + (size_t)(HH + j) * HH;
  const float* vn = W_hh + (size_t)(2 * HH + j) * HH;
  for (int c = 0; c < HH; ++c) {
    float hv = hr[c];
    hrv += hv * vr[c]; hzv += hv * vz[c]; hnv += hv * vn[c];
  }
  float r = 1.f / (1.f + expf(-(ir + hrv)));
  float z = 1.f / (1.f + expf(-(iz + hzv)));
  float n = tanhf(in_ + r * hnv);
  float hn2 = (1.f - z) * n + z * hr[j];
  h_new[idx] = hn2;
  h16[idx] = (_Float16)hn2;
}

// ---------------------------------------------------------------------------
// log_softmax over V=32000 per row + first-occurrence argmax -> y[b]
// ---------------------------------------------------------------------------
__global__ void logsoftmax_argmax_kernel(const float* __restrict__ logits,
                                         float* __restrict__ logp, int* __restrict__ y, int V) {
  const int b = blockIdx.x, tid = threadIdx.x, bs = blockDim.x;
  const float* row = logits + (size_t)b * V;
  __shared__ float sm[256];
  __shared__ int   si[256];

  float m = -3.4e38f; int mi = 0;
  for (int v = tid; v < V; v += bs) {
    float f = row[v];
    if (f > m) { m = f; mi = v; }  // strict > keeps first occurrence in-thread
  }
  sm[tid] = m; si[tid] = mi;
  __syncthreads();
  for (int s = bs >> 1; s; s >>= 1) {
    if (tid < s) {
      if (sm[tid + s] > sm[tid] || (sm[tid + s] == sm[tid] && si[tid + s] < si[tid])) {
        sm[tid] = sm[tid + s]; si[tid] = si[tid + s];
      }
    }
    __syncthreads();
  }
  float rowmax = sm[0];
  if (tid == 0) y[b] = si[0];
  __syncthreads();

  float s = 0.f;
  for (int v = tid; v < V; v += bs) s += expf(row[v] - rowmax);
  sm[tid] = s;
  __syncthreads();
  for (int st = bs >> 1; st; st >>= 1) {
    if (tid < st) sm[tid] += sm[tid + st];
    __syncthreads();
  }
  float logZ = rowmax + logf(sm[0]);
  float* out = logp + (size_t)b * V;
  for (int v = tid; v < V; v += bs) out[v] = row[v] - logZ;
}

// ---------------------------------------------------------------------------
// host launcher
// ---------------------------------------------------------------------------
extern "C" void kernel_launch(void* const* d_in, const int* in_sizes, int n_in,
                              void* d_out, int out_size, void* d_ws, size_t ws_size,
                              hipStream_t stream) {
  const float* features = (const float*)d_in[0];   // [B,K,D]
  const float* emb_W    = (const float*)d_in[1];   // [V,E]
  const float* attn_Wk  = (const float*)d_in[2];   // [D,H]
  const float* attn_Wq  = (const float*)d_in[3];   // [H,H]
  const float* attn_v   = (const float*)d_in[4];   // [H]
  const float* W_ih     = (const float*)d_in[5];   // [3H, E+D]
  const float* W_hh     = (const float*)d_in[6];   // [3H, H]
  const float* b_ih     = (const float*)d_in[7];   // [3H]
  const float* b_hh     = (const float*)d_in[8];   // [3H]
  const float* out_W    = (const float*)d_in[9];   // [V,H]
  const float* out_b    = (const float*)d_in[10];  // [V]
  // d_in[11] = max_len (device scalar); loop count must be host-known: T=20.

  char* ws = (char*)d_ws;
  size_t off = 0;
  auto alloc = [&](size_t bytes) -> void* {
    void* p = (void*)(ws + off);
    off += (bytes + 255) & ~(size_t)255;
    return p;
  };
  _Float16* hW    = (_Float16*)alloc((size_t)VV * HH * sizeof(_Float16)); // out_W f16
  _Float16* hWkT  = (_Float16*)alloc((size_t)DD * HH * sizeof(_Float16)); // attn_Wk^T f16
  float*    KWk   = (float*)alloc((size_t)BB * KK * HH * sizeof(float));
  float*    logits= (float*)alloc((size_t)BB * VV * sizeof(float));
  float*    h0    = (float*)alloc((size_t)BB * HH * sizeof(float));
  float*    h1    = (float*)alloc((size_t)BB * HH * sizeof(float));
  _Float16* h16   = (_Float16*)alloc((size_t)BB * HH * sizeof(_Float16));
  float*    xbuf  = (float*)alloc((size_t)BB * (EE + DD) * sizeof(float));
  int*      ybuf  = (int*)alloc((size_t)BB * sizeof(int));

  float* out_logp = (float*)d_out;                          // [T,B,V]
  float* out_attn = out_logp + (size_t)TT * BB * VV;        // [T,B,K]

  // ---- setup ----
  init_state_kernel<<<(BB * HH + 255) / 256, 256, 0, stream>>>(h0, ybuf, BB * HH, BB);
  cvt_f16_kernel<<<2048, 256, 0, stream>>>(out_W, hW, (size_t)VV * HH);
  cvt_f16_t_kernel<<<dim3(HH / 16, DD / 16), dim3(16, 16), 0, stream>>>(attn_Wk, hWkT, DD, HH);

  // KWk = features @ attn_Wk  (NT with transposed Wk; A read directly as fp32)
  gemm_nt_wmma_lds<float><<<dim3(HH / 64, (BB * KK) / 128), 256, 0, stream>>>(
      features, hWkT, nullptr, KWk, BB * KK, HH, DD);

  // ---- decode loop ----
  float* hc = h0;
  float* hn = h1;
  for (int t = 0; t < TT; ++t) {
    attn_step_kernel<<<BB, 256, 0, stream>>>(hc, attn_Wq, KWk, attn_v, features, emb_W,
                                             ybuf, xbuf, out_attn + (size_t)t * BB * KK);
    gru_step_kernel<<<(BB * HH) / 256, 256, 0, stream>>>(xbuf, hc, W_ih, W_hh, b_ih, b_hh,
                                                         hn, h16);
    gemm_nt_wmma_lds<_Float16><<<dim3(VV / 64, BB / 128), 256, 0, stream>>>(
        h16, hW, out_b, logits, BB, VV, HH);
    logsoftmax_argmax_kernel<<<BB, 256, 0, stream>>>(logits, out_logp + (size_t)t * BB * VV,
                                                     ybuf, VV);
    float* tmp = hc; hc = hn; hn = tmp;
  }

  (void)in_sizes; (void)n_in; (void)out_size; (void)ws_size;
}
